// DGMA_54606214201838
// MI455X (gfx1250) — matrix operand order repaired and checked by reading, not compiled, not run on hardware
//
#include <hip/hip_runtime.h>
#include <hip/hip_bf16.h>

// ---------------- problem constants ----------------
constexpr int Bn   = 8;
constexpr int Cc   = 256;
constexpr int Hh   = 128;
constexpr int Ww   = 128;
constexpr int HW   = Hh * Ww;          // 16384 = 2^14
constexpr int MID  = Cc / 2;           // 128
constexpr int RMID = Cc / 4;           // 64
constexpr int KP   = 5;
constexpr float THRv  = 0.1f;
constexpr float SMINv = 0.05f, SMAXv = 0.45f;
constexpr float BETAv = 1.5f;
constexpr float DMAXv = 0.08f;
constexpr float RMINv = 0.03f, RMAXv = 0.4f;
constexpr float BNEPS = 1e-05f;
constexpr float PIv   = 3.14159265358979323846f;
constexpr int HP = Hh + 2, WP = Ww + 2;   // zero-padded plane for 3x3 implicit GEMM

// swizzled-weight block sizes (elements)
constexpr int PW_WBLK = 8 * 32 * 16;        // per mTile: 8 kc-chunks  (8 KB)
constexpr int C3_WBLK = 9 * 4 * 32 * 16;    // per mTile: 9 taps x 4 kc (36 KB)

// ---------------- bf16 helpers ----------------
__device__ __forceinline__ unsigned short f2bf(float f) {
    unsigned int u = __float_as_uint(f);
    u += 0x7FFFu + ((u >> 16) & 1u);
    return (unsigned short)(u >> 16);
}
__device__ __forceinline__ float bf2f(unsigned short h) {
    return __uint_as_float(((unsigned int)h) << 16);
}

// ---------------- vector types ----------------
typedef __attribute__((ext_vector_type(16))) __bf16          v16bf;
typedef __attribute__((ext_vector_type(16))) unsigned short  u16x16;
typedef __attribute__((ext_vector_type(8)))  unsigned short  u16x8;
typedef __attribute__((ext_vector_type(8)))  float           v8f;
typedef __attribute__((ext_vector_type(4)))  unsigned int    u32x4;
typedef __attribute__((ext_vector_type(8)))  int             i32x8;
typedef __attribute__((ext_vector_type(4)))  int             i32x4;

__device__ __forceinline__ v8f wmma_bf16(v16bf a, v16bf b, v8f c) {
    return __builtin_amdgcn_wmma_f32_16x16x32_bf16(
        false, a, false, b, (short)0, c, false, false);
}
__device__ __forceinline__ v16bf frag_pack(u16x8 a, u16x8 b) {
    u16x16 r;
#pragma unroll
    for (int i = 0; i < 8; ++i) { r[i] = a[i]; r[8 + i] = b[i]; }
    return __builtin_bit_cast(v16bf, r);
}
// A fragment from channel-contiguous activations: two contiguous 16B chunks.
__device__ __forceinline__ v16bf frag2(const unsigned short* p) {
    u16x8 a = *reinterpret_cast<const u16x8*>(p);
    u16x8 b = *reinterpret_cast<const u16x8*>(p + 8);
    return frag_pack(a, b);
}

// ---------------- Tensor Data Mover: 1-D bf16 block copy global -> LDS ----------------
// D# per ISA 8.3/8.4: group0 {count=1, lds_addr, global_addr, type=2},
// group1 {data_size=2B, tensor_dim0=n, tensor_dim1=1, tile_dim0=n, tile_dim1=1}.
// Toolchain uses the 6-arg builtin: (g0, g1, g2, g3, g4, cpol).
__device__ __forceinline__ void tdm_load_1d(unsigned lds_off, const void* gptr, unsigned nelem) {
    unsigned long long ga = (unsigned long long)(uintptr_t)gptr;
    u32x4 g0;
    g0[0] = 1u;                                        // count=1, user mode
    g0[1] = lds_off;                                   // LDS byte address
    g0[2] = (unsigned)ga;                              // global addr [31:0]
    g0[3] = (unsigned)((ga >> 32) & 0x1FFFFFFu) | 0x80000000u;  // [56:32] | type=2
    i32x8 g1;
    g1[0] = (int)(1u << 16);                           // data_size = 2 bytes
    g1[1] = (int)((nelem & 0xFFFFu) << 16);            // tensor_dim0 lo16
    g1[2] = (int)(((nelem >> 16) & 0xFFFFu) | (1u << 16)); // tensor_dim0 hi16 | tensor_dim1 lo16=1
    g1[3] = (int)((nelem & 0xFFFFu) << 16);            // tensor_dim1 hi16=0 | tile_dim0=nelem
    g1[4] = 1;                                         // tile_dim1=1, tile_dim2=0
    g1[5] = (int)nelem;                                // tensor_dim0_stride lo32
    g1[6] = 0;
    g1[7] = 0;
    i32x4 z4 = { 0, 0, 0, 0 };
    i32x8 z8 = { 0, 0, 0, 0, 0, 0, 0, 0 };
    __builtin_amdgcn_tensor_load_to_lds(g0, g1, z4, z4, z8, 0);
}

// ---------------- prep: weight swizzles + zero pad buffer ----------------
// WpwS[mt][kc][lane][e] = bf16( hm_pw1[m = mt*16+(lane&15)][c = kc*32+16*(lane>>4)+e] )
__global__ void k_swz_pw(const float* __restrict__ w, unsigned short* __restrict__ dst) {
    int n = (MID / 16) * PW_WBLK;
    for (int i = blockIdx.x * blockDim.x + threadIdx.x; i < n; i += gridDim.x * blockDim.x) {
        int e = i & 15, lane = (i >> 4) & 31, kc = (i >> 9) & 7, mt = i >> 12;
        int m = mt * 16 + (lane & 15);
        int c = kc * 32 + 16 * (lane >> 4) + e;
        dst[i] = f2bf(w[m * Cc + c]);
    }
}
// Wc3S[mt][t][kc][lane][e] = bf16( hm_c3[m][c][ky][kx] ), t=ky*3+kx
__global__ void k_swz_c3(const float* __restrict__ w, unsigned short* __restrict__ dst) {
    int n = (MID / 16) * C3_WBLK;
    for (int i = blockIdx.x * blockDim.x + threadIdx.x; i < n; i += gridDim.x * blockDim.x) {
        int mt = i / C3_WBLK;
        int r  = i % C3_WBLK;
        int t  = r / 2048;
        int r2 = r % 2048;
        int kc = r2 >> 9;
        int lane = (r2 >> 4) & 31;
        int e = r2 & 15;
        int m = mt * 16 + (lane & 15);
        int c = kc * 32 + 16 * (lane >> 4) + e;
        dst[i] = f2bf(w[(m * MID + c) * 9 + t]);
    }
}
__global__ void k_zero_u16(unsigned short* __restrict__ p, size_t n) {
    for (size_t i = blockIdx.x * (size_t)blockDim.x + threadIdx.x; i < n;
         i += (size_t)gridDim.x * blockDim.x) p[i] = 0;
}

// ---------------- depthwise 3x3 -> transposed bf16 [b][p][c] via LDS transpose ----------------
__global__ void k_dwconv_t(const float* __restrict__ x, const float* __restrict__ wdw,
                           unsigned short* __restrict__ hdwT) {
    const int blk = blockIdx.x;                 // Bn*Hh*(Ww/16)
    const int x0 = (blk & 7) * 16;
    const int y  = (blk >> 3) & (Hh - 1);
    const int b  = blk >> 10;
    const int tid = threadIdx.x;                // 256
    __shared__ unsigned short s[16][Cc];
    const int p = tid & 15;
    const int cb = tid >> 4;
#pragma unroll 4
    for (int it = 0; it < 16; ++it) {
        int c = it * 16 + cb;
        const float* xb = x + ((size_t)b * Cc + c) * HW;
        const float* w = wdw + c * 9;
        int xx = x0 + p;
        float acc = 0.f;
#pragma unroll
        for (int ky = 0; ky < 3; ++ky) {
            int yy = y + ky - 1;
            if ((unsigned)yy >= (unsigned)Hh) continue;
#pragma unroll
            for (int kx = 0; kx < 3; ++kx) {
                int xc = xx + kx - 1;
                if ((unsigned)xc < (unsigned)Ww) acc += xb[yy * Ww + xc] * w[ky * 3 + kx];
            }
        }
        s[p][c] = f2bf(acc);
    }
    __syncthreads();
    const size_t obase = ((size_t)b * HW + y * Ww + x0) * Cc;
#pragma unroll 4
    for (int it = 0; it < 16; ++it)
        hdwT[obase + (size_t)it * Cc + tid] = s[it][tid];
}

// ---------------- 1x1 conv C->MID WMMA GEMM (+BN+ReLU) into padded transposed buffer ----------
__global__ void k_pw_gemm(const unsigned short* __restrict__ hdwT,
                          const unsigned short* __restrict__ WpwS,
                          const float* __restrict__ g, const float* __restrict__ bb,
                          unsigned short* __restrict__ h1pT) {
    __shared__ alignas(32) unsigned short sW[PW_WBLK];   // 8 KB
    const int mTile = blockIdx.y, b = blockIdx.z;
    if (threadIdx.x == 0) {
        tdm_load_1d((unsigned)(uintptr_t)(void*)sW,
                    WpwS + (size_t)mTile * PW_WBLK, PW_WBLK);
        __builtin_amdgcn_s_wait_tensorcnt(0);
    }
    __syncthreads();

    const int lane = threadIdx.x & 31, wv = threadIdx.x >> 5;
    const int lp = lane & 15, hi = lane >> 4;
    const int m = mTile * 16 + lp;
    const float sc = g[m] * rsqrtf(1.0f + BNEPS);
    const float bi = bb[m];

    for (int tt = 0; tt < 8; ++tt) {
        const int nTile = (blockIdx.x * 4 + wv) * 8 + tt;   // 0..1023
        const unsigned short* Abase =
            hdwT + ((size_t)b * HW + nTile * 16 + lp) * Cc + 16 * hi;
        v8f acc = {};
#pragma unroll
        for (int kc = 0; kc < 8; ++kc) {
            v16bf a = frag2(Abase + kc * 32);
            u16x8 w0 = *reinterpret_cast<const u16x8*>(&sW[(kc * 32 + lane) * 16]);
            u16x8 w1 = *reinterpret_cast<const u16x8*>(&sW[(kc * 32 + lane) * 16 + 8]);
            acc = wmma_bf16(a, frag_pack(w0, w1), acc);
        }
        const int p0 = nTile * 16;
#pragma unroll
        for (int r = 0; r < 8; ++r) {
            int p = p0 + r + 8 * hi;
            int y = p >> 7, xx = p & 127;
            float v = fmaxf(acc[r] * sc + bi, 0.0f);
            h1pT[(((size_t)b * HP + (y + 1)) * WP + (xx + 1)) * MID + m] = f2bf(v);
        }
    }
}

// ---------------- 3x3 conv MID->MID implicit WMMA GEMM (+BN+ReLU) ----------------
__global__ void k_conv3_gemm(const unsigned short* __restrict__ h1pT,
                             const unsigned short* __restrict__ Wc3S,
                             const float* __restrict__ g, const float* __restrict__ bb,
                             unsigned short* __restrict__ h2T) {
    __shared__ alignas(32) unsigned short sW[C3_WBLK];   // 36 KB
    const int mTile = blockIdx.y, b = blockIdx.z;
    if (threadIdx.x == 0) {
        tdm_load_1d((unsigned)(uintptr_t)(void*)sW,
                    Wc3S + (size_t)mTile * C3_WBLK, C3_WBLK);
        __builtin_amdgcn_s_wait_tensorcnt(0);
    }
    __syncthreads();

    const int lane = threadIdx.x & 31, wv = threadIdx.x >> 5;
    const int lp = lane & 15, hi = lane >> 4;
    const int m = mTile * 16 + lp;
    const float sc = g[m] * rsqrtf(1.0f + BNEPS);
    const float bi = bb[m];

    for (int tt = 0; tt < 8; ++tt) {
        const int nTile = (blockIdx.x * 4 + wv) * 8 + tt;   // 0..1023
        const int p0 = nTile * 16;
        const int y = p0 >> 7, x0 = p0 & 127;
        v8f acc = {};
#pragma unroll
        for (int t = 0; t < 9; ++t) {
            const int ky = t / 3, kx = t % 3;
            const unsigned short* Abase =
                h1pT + (((size_t)b * HP + (y + ky)) * WP + (x0 + lp + kx)) * MID + 16 * hi;
#pragma unroll
            for (int kc = 0; kc < 4; ++kc) {
                v16bf a = frag2(Abase + kc * 32);
                int wb = ((t * 4 + kc) * 32 + lane) * 16;
                u16x8 w0 = *reinterpret_cast<const u16x8*>(&sW[wb]);
                u16x8 w1 = *reinterpret_cast<const u16x8*>(&sW[wb + 8]);
                acc = wmma_bf16(a, frag_pack(w0, w1), acc);
            }
        }
#pragma unroll
        for (int r = 0; r < 8; ++r) {
            int p = p0 + r + 8 * hi;
            float v = fmaxf(acc[r] * sc + bi, 0.0f);
            h2T[((size_t)b * HW + p) * MID + m] = f2bf(v);
        }
    }
}

// ---------------- heat head: per-pixel 128-dot + sigmoid -> out channel 1 ----------------
__global__ void k_head(const unsigned short* __restrict__ h2T,
                       const float* __restrict__ w, const float* __restrict__ bias,
                       float* __restrict__ out) {
    size_t i = blockIdx.x * (size_t)blockDim.x + threadIdx.x;
    if (i >= (size_t)Bn * HW) return;
    int b = (int)(i >> 14);
    int p = (int)(i & (HW - 1));
    const unsigned short* hp = h2T + ((size_t)b * HW + p) * MID;
    float acc = bias[0];
#pragma unroll
    for (int mc = 0; mc < MID; mc += 8) {
        u16x8 v = *reinterpret_cast<const u16x8*>(hp + mc);
#pragma unroll
        for (int j = 0; j < 8; ++j) acc += bf2f(v[j]) * w[mc + j];
    }
    out[((size_t)b * 2 + 1) * HW + p] = 1.0f / (1.0f + expf(-acc));
}

// ---------------- top-K peaks per batch ----------------
__global__ void k_topk(const float* __restrict__ out, float* __restrict__ params) {
    const int b = blockIdx.x, tid = threadIdx.x;   // 256 threads
    const float* heat = out + ((size_t)b * 2 + 1) * HW;
    float tv[KP]; int ti[KP];
#pragma unroll
    for (int j = 0; j < KP; ++j) { tv[j] = -1.0f; ti[j] = -1; }

    for (int p = tid; p < HW; p += 256) {
        int y = p >> 7, xx = p & 127;
        float v = heat[p];
        float mx = v;
#pragma unroll
        for (int ky = -1; ky <= 1; ++ky) {
            int yy = y + ky;
            if ((unsigned)yy >= (unsigned)Hh) continue;
#pragma unroll
            for (int kx = -1; kx <= 1; ++kx) {
                int xc = xx + kx;
                if ((unsigned)xc < (unsigned)Ww) mx = fmaxf(mx, heat[yy * Ww + xc]);
            }
        }
        float pk = (v >= mx) ? v : 0.0f;
        if (pk > tv[KP - 1]) {
            int j = KP - 1;
            while (j > 0 && pk > tv[j - 1]) { tv[j] = tv[j - 1]; ti[j] = ti[j - 1]; --j; }
            tv[j] = pk; ti[j] = p;
        }
    }

    __shared__ float sv[256];
    __shared__ int   si[256];
    __shared__ float bestv[KP];
    __shared__ int   besti[KP];
    for (int r = 0; r < KP; ++r) {
        sv[tid] = tv[0]; si[tid] = ti[0];
        __syncthreads();
        for (int s = 128; s > 0; s >>= 1) {
            if (tid < s && sv[tid + s] > sv[tid]) { sv[tid] = sv[tid + s]; si[tid] = si[tid + s]; }
            __syncthreads();
        }
        if (tid == 0) { bestv[r] = sv[0]; besti[r] = si[0]; }
        __syncthreads();
        if (ti[0] == besti[r] && besti[r] >= 0) {
#pragma unroll
            for (int j = 0; j < KP - 1; ++j) { tv[j] = tv[j + 1]; ti[j] = ti[j + 1]; }
            tv[KP - 1] = -1.0f; ti[KP - 1] = -1;
        }
        __syncthreads();
    }
    if (tid == 0) {
        for (int r = 0; r < KP; ++r) {
            float v = bestv[r];
            int idx = besti[r] < 0 ? 0 : besti[r];
            float valid = (besti[r] >= 0 && v >= THRv) ? 1.0f : 0.0f;
            int row = idx >> 7, col = idx & 127;
            float ny = 2.0f * row / (Hh - 1) - 1.0f;
            float nx = 2.0f * col / (Ww - 1) - 1.0f;
            float* pp = params + ((size_t)b * KP + r) * 8;
            pp[0] = nx * valid; pp[1] = ny * valid; pp[2] = valid;
        }
    }
}

// ---------------- per-peak sampling + on-the-fly radius branch + MLP ----------------
__global__ void k_mlp(const float* __restrict__ x,
                      const float* __restrict__ r_dw, const float* __restrict__ r_pw1,
                      const float* __restrict__ r_g,  const float* __restrict__ r_b,
                      const float* __restrict__ r_out_w, const float* __restrict__ r_out_b,
                      const float* __restrict__ log_alpha,
                      const float* __restrict__ w1, const float* __restrict__ b1,
                      const float* __restrict__ w2, const float* __restrict__ b2,
                      float* __restrict__ params) {
    const int b = blockIdx.x, tid = threadIdx.x;   // 256 threads
    __shared__ float sfeat[Cc];
    __shared__ float srdw[4][Cc];
    __shared__ float sr[4][RMID];
    __shared__ float sp1[MID];
    __shared__ float sp[4];

    for (int k = 0; k < KP; ++k) {
        float* pp = params + ((size_t)b * KP + k) * 8;
        const float cx = pp[0], cy = pp[1];
        float px = fminf(fmaxf((cx + 1.0f) * 0.5f * (Ww - 1), 0.0f), (float)(Ww - 1));
        float py = fminf(fmaxf((cy + 1.0f) * 0.5f * (Hh - 1), 0.0f), (float)(Hh - 1));
        int x0 = (int)floorf(px), y0 = (int)floorf(py);
        int x1 = min(x0 + 1, Ww - 1), y1 = min(y0 + 1, Hh - 1);
        float wx = px - x0, wy = py - y0;

        const int c = tid;
        const float* xb = x + ((size_t)b * Cc + c) * HW;
        float v00 = xb[y0 * Ww + x0], v01 = xb[y0 * Ww + x1];
        float v10 = xb[y1 * Ww + x0], v11 = xb[y1 * Ww + x1];
        sfeat[c] = (1 - wy) * ((1 - wx) * v00 + wx * v01) + wy * ((1 - wx) * v10 + wx * v11);

        const float* wd = r_dw + c * 9;
        int pys[4] = { y0, y0, y1, y1 };
        int pxs[4] = { x0, x1, x0, x1 };
#pragma unroll
        for (int q = 0; q < 4; ++q) {
            int yq = pys[q], xq = pxs[q];
            float a = 0.f;
#pragma unroll
            for (int ky = 0; ky < 3; ++ky) {
                int yy = yq + ky - 1;
                if ((unsigned)yy >= (unsigned)Hh) continue;
#pragma unroll
                for (int kx = 0; kx < 3; ++kx) {
                    int xc = xq + kx - 1;
                    if ((unsigned)xc < (unsigned)Ww) a += xb[yy * Ww + xc] * wd[ky * 3 + kx];
                }
            }
            srdw[q][c] = a;
        }
        __syncthreads();

        if (tid < 4 * RMID) {
            int q = tid >> 6, j = tid & 63;
            const float* wp = r_pw1 + (size_t)j * Cc;
            float a = 0.f;
            for (int c2 = 0; c2 < Cc; ++c2) a += wp[c2] * srdw[q][c2];
            a = a * (r_g[j] * rsqrtf(1.0f + BNEPS)) + r_b[j];
            sr[q][j] = fmaxf(a, 0.0f);
        }
        __syncthreads();
        if (tid < 4) {
            float a = r_out_b[0];
            for (int j = 0; j < RMID; ++j) a += r_out_w[j] * sr[tid][j];
            sp[tid] = RMINv + (1.0f / (1.0f + expf(-a))) * (RMAXv - RMINv);
        }
        __syncthreads();
        float r_k = (1 - wy) * ((1 - wx) * sp[0] + wx * sp[1]) +
                    wy * ((1 - wx) * sp[2] + wx * sp[3]);
        __syncthreads();

        if (tid < MID) {
            float a = b1[tid];
            for (int c2 = 0; c2 < Cc; ++c2) a += sfeat[c2] * w1[(size_t)c2 * MID + tid];
            sp1[tid] = fmaxf(a, 0.0f);
        }
        __syncthreads();
        if (tid < 4) {
            float a = b2[tid];
            for (int j = 0; j < MID; ++j) a += sp1[j] * w2[(size_t)j * 4 + tid];
            sp[tid] = a;
        }
        __syncthreads();
        if (tid == 0) {
            float alpha = log1pf(expf(log_alpha[0]));
            float dsx = tanhf(sp[0]) * DMAXv;
            float dsy = tanhf(sp[1]) * DMAXv;
            float theta = tanhf(sp[2]) * PIv;
            float wgt = 1.0f / (1.0f + expf(-sp[3]));
            float sx = fminf(fmaxf(alpha * r_k + dsx, SMINv), SMAXv);
            float sy = fminf(fmaxf(alpha * r_k * BETAv + dsy, SMINv), SMAXv);
            pp[3] = wgt; pp[4] = sx; pp[5] = sy; pp[6] = cosf(theta); pp[7] = sinf(theta);
        }
        __syncthreads();
    }
}

// ---------------- Gaussian mixture attention -> out channel 0 ----------------
__global__ void k_attn(const float* __restrict__ params, float* __restrict__ out) {
    size_t i = blockIdx.x * (size_t)blockDim.x + threadIdx.x;
    if (i >= (size_t)Bn * HW) return;
    int b = (int)(i >> 14);
    int p = (int)(i & (HW - 1));
    int y = p >> 7, xx = p & 127;
    float gx = 2.0f * xx / (Ww - 1) - 1.0f;
    float gy = 2.0f * y  / (Hh - 1) - 1.0f;
    float wsum = 0.f, num = 0.f;
#pragma unroll
    for (int k = 0; k < KP; ++k) {
        const float* pp = params + ((size_t)b * KP + k) * 8;
        float cx = pp[0], cy = pp[1], valid = pp[2], wgt = pp[3];
        float sx = pp[4], sy = pp[5], ct = pp[6], st = pp[7];
        float dx = gx - cx, dy = gy - cy;
        float xr = ct * dx + st * dy;
        float yr = -st * dx + ct * dy;
        float G = expf(-(xr * xr / (2.f * sx * sx + 1e-6f) +
                         yr * yr / (2.f * sy * sy + 1e-6f)));
        float mw = wgt * valid;
        wsum += mw;
        num  += G * mw * valid;
    }
    wsum = fmaxf(wsum, 1e-6f);
    float mix = num / wsum;
    out[((size_t)b * 2) * HW + p] = 1.0f / (1.0f + expf(-(mix * 4.0f - 2.0f)));
}

// ---------------- launch ----------------
extern "C" void kernel_launch(void* const* d_in, const int* in_sizes, int n_in,
                              void* d_out, int out_size, void* d_ws, size_t ws_size,
                              hipStream_t stream) {
    const float* x        = (const float*)d_in[0];
    const float* hm_dw    = (const float*)d_in[1];
    const float* hm_pw1   = (const float*)d_in[2];
    const float* hm_g1    = (const float*)d_in[3];
    const float* hm_b1    = (const float*)d_in[4];
    const float* hm_c3    = (const float*)d_in[5];
    const float* hm_g2    = (const float*)d_in[6];
    const float* hm_b2    = (const float*)d_in[7];
    const float* hm_out_w = (const float*)d_in[8];
    const float* hm_out_b = (const float*)d_in[9];
    const float* r_dw     = (const float*)d_in[10];
    const float* r_pw1    = (const float*)d_in[11];
    const float* r_g      = (const float*)d_in[12];
    const float* r_b      = (const float*)d_in[13];
    const float* r_out_w  = (const float*)d_in[14];
    const float* r_out_b  = (const float*)d_in[15];
    const float* log_a    = (const float*)d_in[16];
    const float* mlp_w1   = (const float*)d_in[17];
    const float* mlp_b1   = (const float*)d_in[18];
    const float* mlp_w2   = (const float*)d_in[19];
    const float* mlp_b2   = (const float*)d_in[20];
    float* out = (float*)d_out;

    char* ws = (char*)d_ws;
    size_t off = 0;
    auto take = [&](size_t bytes) { size_t o = off; off += (bytes + 255) & ~(size_t)255; return o; };
    unsigned short* WpwS   = (unsigned short*)(ws + take((size_t)(MID / 16) * PW_WBLK * 2));
    unsigned short* Wc3S   = (unsigned short*)(ws + take((size_t)(MID / 16) * C3_WBLK * 2));
    unsigned short* hdwT   = (unsigned short*)(ws + take((size_t)Bn * HW * Cc * 2));
    unsigned short* h1pT   = (unsigned short*)(ws + take((size_t)Bn * HP * WP * MID * 2));
    unsigned short* h2T    = (unsigned short*)(ws + take((size_t)Bn * HW * MID * 2));
    float*          params = (float*)(ws + take((size_t)Bn * KP * 8 * 4));
    (void)in_sizes; (void)n_in; (void)out_size; (void)ws_size;

    // 1. weight swizzles + zero padded buffer
    k_swz_pw<<<128, 256, 0, stream>>>(hm_pw1, WpwS);
    k_swz_c3<<<288, 256, 0, stream>>>(hm_c3, Wc3S);
    k_zero_u16<<<4096, 256, 0, stream>>>(h1pT, (size_t)Bn * HP * WP * MID);

    // 2. depthwise 3x3 -> transposed bf16
    k_dwconv_t<<<Bn * Hh * (Ww / 16), 256, 0, stream>>>(x, hm_dw, hdwT);

    // 3. 1x1 C->MID WMMA GEMM (TDM-staged weights)
    k_pw_gemm<<<dim3(32, MID / 16, Bn), 128, 0, stream>>>(hdwT, WpwS, hm_g1, hm_b1, h1pT);

    // 4. 3x3 MID->MID implicit WMMA GEMM (TDM-staged weights)
    k_conv3_gemm<<<dim3(32, MID / 16, Bn), 128, 0, stream>>>(h1pT, Wc3S, hm_g2, hm_b2, h2T);

    // 5. heat head -> out channel 1
    k_head<<<(Bn * HW) / 256, 256, 0, stream>>>(h2T, hm_out_w, hm_out_b, out);

    // 6. top-K peaks -> params[0..2]
    k_topk<<<Bn, 256, 0, stream>>>(out, params);

    // 7. per-peak sampling + on-the-fly radius branch + MLP -> params[3..7]
    k_mlp<<<Bn, 256, 0, stream>>>(x, r_dw, r_pw1, r_g, r_b, r_out_w, r_out_b, log_a,
                                  mlp_w1, mlp_b1, mlp_w2, mlp_b2, params);

    // 8. Gaussian mixture attention -> out channel 0
    k_attn<<<(Bn * HW) / 256, 256, 0, stream>>>(params, out);
}
